// PGSFNet_8100308321060
// MI455X (gfx1250) — matrix operand order repaired and checked by reading, not compiled, not run on hardware
//
#include <hip/hip_runtime.h>
#include <cstdint>
#include <cstddef>

typedef _Float16 h16;
typedef __attribute__((ext_vector_type(16))) _Float16 v16h;
typedef __attribute__((ext_vector_type(8)))  _Float16 v8h;
typedef __attribute__((ext_vector_type(8)))  float    v8f;
typedef __attribute__((ext_vector_type(4)))  unsigned int u32x4;
typedef __attribute__((ext_vector_type(8)))  unsigned int u32x8;

#define BB_  2
#define HH_  192
#define WW_  192
#define CC_  192
#define CH_  96
#define TT_  (BB_*HH_*WW_)     // 73728 tokens
#define WSZ  8
#define NWY  (HH_/WSZ)         // 24
#define NWX  (WW_/WSZ)         // 24
#define NWIN (BB_*NWY*NWX)     // 1152 windows

// ---- WMMA GEMM tiling ----
#define BM 128
#define BN 96
#define BK 32
#define LSTR 40               // LDS row stride (halves): 64B row + 16B TDM pad

enum { EPI_F16 = 0, EPI_GELU_F16 = 1, EPI_RES_F32 = 2 };

// ---------------------------------------------------------------------------
// Fragment loaders (CDNA5 wave32 WMMA 16x16x32 f16 layouts, ISA 7.12.2)
// ---------------------------------------------------------------------------
__device__ __forceinline__ v16h frag_a(const h16* base) {
  const int l = threadIdx.x & 31;
  const h16* p = base + (l & 15) * LSTR + ((l < 16) ? 0 : 8);
  v8h lo = *(const v8h*)p;
  v8h hi = *(const v8h*)(p + 16);
  v16h f;
#pragma unroll
  for (int i = 0; i < 8; ++i) { f[i] = lo[i]; f[i + 8] = hi[i]; }
  return f;
}
__device__ __forceinline__ v16h frag_b(const h16* base) {
  const int l = threadIdx.x & 31;
  const h16* p = base + (l & 15) * LSTR + ((l < 16) ? 0 : 8);
  v8h lo = *(const v8h*)p;
  v8h hi = *(const v8h*)(p + 16);
  v16h f;
#pragma unroll
  for (int i = 0; i < 8; ++i) { f[i] = lo[i]; f[i + 8] = hi[i]; }
  return f;
}

__device__ __forceinline__ float gelu_f(float x) {
  return 0.5f * x * (1.f + tanhf(0.7978845608028654f * (x + 0.044715f * x * x * x)));
}

// LDS byte offset of a __shared__ object (generic ptr low 32 bits = LDS addr)
__device__ __forceinline__ uint32_t lds_off_of(const void* p) {
  return (uint32_t)(uintptr_t)p;
}

// ---------------------------------------------------------------------------
// Async gather of 16B per lane directly into LDS (ASYNCcnt).
// IOFFSET applies to BOTH lds and global addresses (ISA 15.18.3).
// ---------------------------------------------------------------------------
__device__ __forceinline__ void async_ld_lds_b128(uint32_t lds_addr, const void* gaddr) {
  asm volatile("global_load_async_to_lds_b128 %0, %1, off"
               :: "v"(lds_addr), "v"(gaddr) : "memory");
}
__device__ __forceinline__ void async_ld_lds_b128_o16(uint32_t lds_addr, const void* gaddr) {
  asm volatile("global_load_async_to_lds_b128 %0, %1, off offset:16"
               :: "v"(lds_addr), "v"(gaddr) : "memory");
}
__device__ __forceinline__ void wait_asynccnt0() {
  asm volatile("s_wait_asynccnt 0x0" ::: "memory");
}

// ---------------------------------------------------------------------------
// Tensor Data Mover: load a [rows x 32] f16 tile (row stride = K elements)
// into LDS with 16B pad after every 64B row  ->  LDS row stride = 40 halves.
// D# per CDNA5 ISA. 2-D tile: groups 0+1 only.
// Issue from ONE wave (EXEC ignored by TDM; wave-level instruction).
// ---------------------------------------------------------------------------
__device__ __forceinline__ void tdm_load_b_tile(const h16* gsrc, uint32_t lds_byte_off,
                                                int K, int rows) {
  const uint64_t ga = (uint64_t)(uintptr_t)gsrc;
  u32x4 g0;
  g0[0] = 1u;                                               // count=1, user D#
  g0[1] = lds_byte_off;                                     // lds_addr
  g0[2] = (uint32_t)ga;                                     // global_addr[31:0]
  g0[3] = (uint32_t)((ga >> 32) & 0x1FFFFFFu) | (2u << 30); // addr[56:32] | type=2
  u32x8 g1;
  g1[0] = (1u << 16)            // data_size = 2B
        | (1u << 20)            // pad_enable
        | (3u << 22)            // pad_interval: every 16 DWORDs (64B)
        | (3u << 25);           // pad_amount: 4 DWORDs (16B)
  g1[1] = ((uint32_t)K & 0xFFFFu) << 16;                    // tensor_dim0[15:0]
  g1[2] = ((uint32_t)K >> 16) | ((uint32_t)rows << 16);     // tensor_dim0 hi | tensor_dim1 lo
  g1[3] = (32u << 16);                                      // tensor_dim1 hi=0 | tile_dim0=32
  g1[4] = (uint32_t)rows;                                   // tile_dim1 | tile_dim2=0
  g1[5] = (uint32_t)K;                                      // tensor_dim0_stride[31:0]
  g1[6] = 0u;                                               // stride0 hi | stride1 lo
  g1[7] = 0u;                                               // stride1 hi
  asm volatile("tensor_load_to_lds %0, %1" :: "s"(g0), "s"(g1) : "memory");
}

// ---------------------------------------------------------------------------
// Generic WMMA GEMM: out[M,N] = epilogue(A[M,K] @ Bt[N,K]^T + bias)
// block = 256 thr (8 waves); block tile 128x96; wave tile 32x48 (2x3 frags)
// A tile via per-lane async-to-LDS; B tile via TDM with padding.
// ---------------------------------------------------------------------------
__global__ __launch_bounds__(256) void gemm_wmma(
    const h16* __restrict__ A, int lda,
    const h16* __restrict__ Bt, int K,
    const float* __restrict__ bias,
    const float* __restrict__ resid,
    h16* __restrict__ out_h, float* __restrict__ out_f,
    int ldo, int mode)
{
  __shared__ h16 sA[BM * LSTR];
  __shared__ h16 sB[BN * LSTR];
  const int m0 = blockIdx.x * BM;
  const int n0 = blockIdx.y * BN;
  const int t = threadIdx.x;
  const int lane = t & 31;
  const int wid = t >> 5;
  const int wm = wid & 3;   // 4 waves in M
  const int wn = wid >> 2;  // 2 waves in N

  v8f acc[2][3];
#pragma unroll
  for (int f = 0; f < 2; ++f)
#pragma unroll
    for (int g = 0; g < 3; ++g)
#pragma unroll
      for (int i = 0; i < 8; ++i) acc[f][g][i] = 0.f;

  const int arow = t >> 1;
  const int ac = (t & 1) * 16;
  const uint32_t sA_dst = lds_off_of(sA) + (uint32_t)(arow * LSTR + ac) * 2u;
  const uint32_t sB_base = lds_off_of(sB);

  for (int k0 = 0; k0 < K; k0 += BK) {
    const h16* asrc = A + (size_t)(m0 + arow) * lda + k0 + ac;
    async_ld_lds_b128(sA_dst, asrc);          // halves [ac, ac+8)
    async_ld_lds_b128_o16(sA_dst, asrc);      // halves [ac+8, ac+16)
    if (wid == 0)
      tdm_load_b_tile(Bt + (size_t)n0 * K + k0, sB_base, K, BN);
    if (k0 + BK < K) __builtin_prefetch(asrc + BK, 0, 3);
    wait_asynccnt0();
    __builtin_amdgcn_s_wait_tensorcnt(0);
    __syncthreads();

    v16h bf[3];
#pragma unroll
    for (int g = 0; g < 3; ++g) bf[g] = frag_b(sB + (wn * 48 + g * 16) * LSTR);
#pragma unroll
    for (int f = 0; f < 2; ++f) {
      v16h af = frag_a(sA + (wm * 32 + f * 16) * LSTR);
#pragma unroll
      for (int g = 0; g < 3; ++g)
        acc[f][g] = __builtin_amdgcn_wmma_f32_16x16x32_f16(
            false, af, false, bf[g], (short)0, acc[f][g], false, false);
    }
    __syncthreads();
  }

  // epilogue: C/D layout lane l -> col (l&15), row v + (l>=16 ? 8 : 0)
  const int nl = lane & 15;
  const int mo = (lane >> 4) << 3;
#pragma unroll
  for (int f = 0; f < 2; ++f) {
    const int mbase = m0 + wm * 32 + f * 16 + mo;
#pragma unroll
    for (int g = 0; g < 3; ++g) {
      const int n = n0 + wn * 48 + g * 16 + nl;
      const float bv = bias ? bias[n] : 0.f;
#pragma unroll
      for (int v = 0; v < 8; ++v) {
        const size_t idx = (size_t)(mbase + v) * ldo + n;
        float val = acc[f][g][v] + bv;
        if (mode == EPI_F16) {
          out_h[idx] = (h16)val;
        } else if (mode == EPI_GELU_F16) {
          out_h[idx] = (h16)gelu_f(val);
        } else {  // EPI_RES_F32
          out_f[idx] = resid[idx] + val;
        }
      }
    }
  }
}

// ---------------------------------------------------------------------------
// 3x3 SAME conv as implicit GEMM: M=pixels, N=192 (c_out), K=9*192=1728.
// ---------------------------------------------------------------------------
__global__ __launch_bounds__(256) void conv_gemm(
    const h16* __restrict__ xc,          // [T][192] f16
    const h16* __restrict__ Bt,          // [192][1728] f16 (c_out major)
    const float* __restrict__ bias,      // [192]
    const float* __restrict__ res0,      // original input x (f32)
    float* __restrict__ out)             // d_out f32 [T][192]
{
  __shared__ h16 sA[BM * LSTR];
  __shared__ h16 sB[BN * LSTR];
  const int m0 = blockIdx.x * BM;
  const int n0 = blockIdx.y * BN;
  const int t = threadIdx.x;
  const int lane = t & 31;
  const int wid = t >> 5;
  const int wm = wid & 3;
  const int wn = wid >> 2;
  const int K = 9 * CC_;

  v8f acc[2][3];
#pragma unroll
  for (int f = 0; f < 2; ++f)
#pragma unroll
    for (int g = 0; g < 3; ++g)
#pragma unroll
      for (int i = 0; i < 8; ++i) acc[f][g][i] = 0.f;

  const int arow = t >> 1;
  const int ac = (t & 1) * 16;
  const int p = m0 + arow;
  const int bimg = p / (HH_ * WW_);
  const int rem = p % (HH_ * WW_);
  const int py = rem / WW_;
  const int px = rem % WW_;
  const uint32_t sA_dst = lds_off_of(sA) + (uint32_t)(arow * LSTR + ac) * 2u;
  const uint32_t sB_base = lds_off_of(sB);

  for (int k0 = 0; k0 < K; k0 += BK) {
    const int tap = k0 / CC_;
    const int dy = tap / 3 - 1, dx = tap % 3 - 1;
    const int cin = (k0 % CC_) + ac;
    const int sy = py + dy, sx = px + dx;
    if (sy >= 0 && sy < HH_ && sx >= 0 && sx < WW_) {
      const h16* asrc = xc + ((size_t)bimg * HH_ * WW_ + (size_t)sy * WW_ + sx) * CC_ + cin;
      async_ld_lds_b128(sA_dst, asrc);
      async_ld_lds_b128_o16(sA_dst, asrc);
    } else {
      h16* adst = sA + arow * LSTR + ac;
      float4 z = make_float4(0.f, 0.f, 0.f, 0.f);
      *(float4*)adst = z;
      *(float4*)(adst + 8) = z;
    }
    if (wid == 0)
      tdm_load_b_tile(Bt + (size_t)n0 * K + k0, sB_base, K, BN);
    wait_asynccnt0();
    __builtin_amdgcn_s_wait_tensorcnt(0);
    __syncthreads();

    v16h bf[3];
#pragma unroll
    for (int g = 0; g < 3; ++g) bf[g] = frag_b(sB + (wn * 48 + g * 16) * LSTR);
#pragma unroll
    for (int f = 0; f < 2; ++f) {
      v16h af = frag_a(sA + (wm * 32 + f * 16) * LSTR);
#pragma unroll
      for (int g = 0; g < 3; ++g)
        acc[f][g] = __builtin_amdgcn_wmma_f32_16x16x32_f16(
            false, af, false, bf[g], (short)0, acc[f][g], false, false);
    }
    __syncthreads();
  }

  const int nl = lane & 15;
  const int mo = (lane >> 4) << 3;
#pragma unroll
  for (int f = 0; f < 2; ++f) {
    const int mbase = m0 + wm * 32 + f * 16 + mo;
#pragma unroll
    for (int g = 0; g < 3; ++g) {
      const int n = n0 + wn * 48 + g * 16 + nl;
      const float bv = bias[n];
#pragma unroll
      for (int v = 0; v < 8; ++v) {
        const size_t idx = (size_t)(mbase + v) * CC_ + n;
        out[idx] = res0[idx] + acc[f][g][v] + bv;
      }
    }
  }
}

// ---------------------------------------------------------------------------
// LayerNorm over C=192 (one token per block, 192 threads) -> f16 output
// ---------------------------------------------------------------------------
__global__ __launch_bounds__(192) void ln_kernel(
    const float* __restrict__ x, const float* __restrict__ g,
    const float* __restrict__ b, h16* __restrict__ xn)
{
  __shared__ float s1[CC_], s2[CC_];
  const int tok = blockIdx.x;
  const int c = threadIdx.x;
  const float v = x[(size_t)tok * CC_ + c];
  s1[c] = v; s2[c] = v * v;
  __syncthreads();
  for (int s = 96; s >= 3; s >>= 1) {
    if (c < s) { s1[c] += s1[c + s]; s2[c] += s2[c + s]; }
    __syncthreads();
  }
  if (c == 0) { s1[0] = s1[0] + s1[1] + s1[2]; s2[0] = s2[0] + s2[1] + s2[2]; }
  __syncthreads();
  const float mu = s1[0] * (1.f / CC_);
  const float var = s2[0] * (1.f / CC_) - mu * mu;
  const float rs = rsqrtf(var + 1e-5f);
  xn[(size_t)tok * CC_ + c] = (h16)((v - mu) * rs * g[c] + b[c]);
}

// ---------------------------------------------------------------------------
// Window attention (first 96 channels), per 8x8 window, online softmax.
// ---------------------------------------------------------------------------
__global__ __launch_bounds__(256) void win_attn(
    const h16* __restrict__ qkv,    // [T][288]
    const float* __restrict__ rpb,  // [225][4] f32
    h16* __restrict__ out,          // [T][192], cols 0..95
    int shift)
{
  __shared__ h16 tile[64 * 288];
  __shared__ float rp[225 * 4];
  const int blk = blockIdx.x;
  const int bimg = blk / (NWY * NWX);
  const int wrem = blk % (NWY * NWX);
  const int wh = wrem / NWX, ww = wrem % NWX;
  const int t = threadIdx.x;
  const size_t tokbase = (size_t)bimg * HH_ * WW_;

  for (int c = t; c < 64 * 288 / 8; c += 256) {
    const int row = c / 36;
    const int off = (c % 36) * 8;
    const int r = row >> 3, cl = row & 7;
    const int gy = (wh * WSZ + r + shift) % HH_;
    const int gx = (ww * WSZ + cl + shift) % WW_;
    const size_t tok = tokbase + (size_t)gy * WW_ + gx;
    *(v8h*)&tile[row * 288 + off] = *(const v8h*)&qkv[tok * 288 + off];
  }
  for (int c = t; c < 225 * 4; c += 256) rp[c] = rpb[c];
  __syncthreads();

  const int hd = t >> 6, n = t & 63;
  const int rn = n >> 3, cn = n & 7;
  const float scale = 0.2041241452319315f;  // 1/sqrt(24)
  const h16* qp = &tile[n * 288 + hd * 24];

  int regn = 0;
  if (shift) {
    const int gy = wh * WSZ + rn, gx = ww * WSZ + cn;
    const int ry = gy >= HH_ - 4 ? 2 : (gy >= HH_ - 8 ? 1 : 0);
    const int rx = gx >= WW_ - 4 ? 2 : (gx >= WW_ - 8 ? 1 : 0);
    regn = ry * 3 + rx;
  }

  float mx = -1e30f, sum = 0.f;
  float od[24];
#pragma unroll
  for (int d = 0; d < 24; ++d) od[d] = 0.f;

  for (int m = 0; m < 64; ++m) {
    const h16* kp = &tile[m * 288 + 96 + hd * 24];
    float s = 0.f;
#pragma unroll
    for (int d = 0; d < 24; ++d) s += (float)qp[d] * (float)kp[d];
    s *= scale;
    const int rm = m >> 3, cm = m & 7;
    s += rp[((rn - rm + 7) * 15 + (cn - cm + 7)) * 4 + hd];
    if (shift) {
      const int gy = wh * WSZ + rm, gx = ww * WSZ + cm;
      const int ry = gy >= HH_ - 4 ? 2 : (gy >= HH_ - 8 ? 1 : 0);
      const int rx = gx >= WW_ - 4 ? 2 : (gx >= WW_ - 8 ? 1 : 0);
      if (ry * 3 + rx != regn) s -= 100.f;
    }
    if (s > mx) {
      const float fct = __expf(mx - s);
      sum *= fct;
#pragma unroll
      for (int d = 0; d < 24; ++d) od[d] *= fct;
      mx = s;
    }
    const float pr = __expf(s - mx);
    sum += pr;
    const h16* vp = &tile[m * 288 + 192 + hd * 24];
#pragma unroll
    for (int d = 0; d < 24; ++d) od[d] += pr * (float)vp[d];
  }
  const float inv = 1.f / sum;
  const int gy = (wh * WSZ + rn + shift) % HH_;
  const int gx = (ww * WSZ + cn + shift) % WW_;
  const size_t tok = tokbase + (size_t)gy * WW_ + gx;
#pragma unroll
  for (int d = 0; d < 24; ++d)
    out[tok * CC_ + hd * 24 + d] = (h16)(od[d] * inv);
}

// ---------------------------------------------------------------------------
// Anchored stripe attention (second 96 channels), per 8x8 window, 16 anchors.
// ---------------------------------------------------------------------------
__global__ __launch_bounds__(256) void stripe_attn(
    const h16* __restrict__ qkv,  // [T][288]
    const h16* __restrict__ xn,   // [T][192] (anchor source, cols 96..191)
    h16* __restrict__ out)        // [T][192], cols 96..191
{
  __shared__ h16 tile[64 * 288];
  __shared__ float anc[16 * 96];   // [cell][head*24+d]
  __shared__ float tmat[4 * 16 * 24];
  const int blk = blockIdx.x;
  const int bimg = blk / (NWY * NWX);
  const int wrem = blk % (NWY * NWX);
  const int wh = wrem / NWX, ww = wrem % NWX;
  const int t = threadIdx.x;
  const size_t tokbase = (size_t)bimg * HH_ * WW_;
  const float scale = 0.2041241452319315f;

  for (int c = t; c < 64 * 288 / 8; c += 256) {
    const int row = c / 36;
    const int off = (c % 36) * 8;
    const int r = row >> 3, cl = row & 7;
    const size_t tok = tokbase + (size_t)(wh * WSZ + r) * WW_ + (ww * WSZ + cl);
    *(v8h*)&tile[row * 288 + off] = *(const v8h*)&qkv[tok * 288 + off];
  }
  for (int idx = t; idx < 16 * 96; idx += 256) {
    const int cell = idx / 96, ch = idx % 96;
    const int ci = cell >> 2, cj = cell & 3;
    float sm = 0.f;
#pragma unroll
    for (int a = 0; a < 2; ++a)
#pragma unroll
      for (int bq = 0; bq < 2; ++bq) {
        const size_t tok = tokbase + (size_t)(wh * WSZ + ci * 2 + a) * WW_ + (ww * WSZ + cj * 2 + bq);
        sm += (float)xn[tok * CC_ + 96 + ch];
      }
    anc[idx] = sm * 0.25f;
  }
  __syncthreads();

  if (t < 64) {
    const int hd = t >> 4, m = t & 15;
    const float* ap = &anc[m * 96 + hd * 24];
    float a24[24];
#pragma unroll
    for (int d = 0; d < 24; ++d) a24[d] = 0.f;
    float mx = -1e30f, sum = 0.f;
    for (int n = 0; n < 64; ++n) {
      const h16* kp = &tile[n * 288 + 96 + hd * 24];
      float s = 0.f;
#pragma unroll
      for (int d = 0; d < 24; ++d) s += ap[d] * (float)kp[d];
      s *= scale;
      if (s > mx) {
        const float fct = __expf(mx - s);
        sum *= fct;
#pragma unroll
        for (int d = 0; d < 24; ++d) a24[d] *= fct;
        mx = s;
      }
      const float pr = __expf(s - mx);
      sum += pr;
      const h16* vp = &tile[n * 288 + 192 + hd * 24];
#pragma unroll
      for (int d = 0; d < 24; ++d) a24[d] += pr * (float)vp[d];
    }
    const float inv = 1.f / sum;
#pragma unroll
    for (int d = 0; d < 24; ++d) tmat[(hd * 16 + m) * 24 + d] = a24[d] * inv;
  }
  __syncthreads();

  const int hd = t >> 6, n = t & 63;
  const h16* qp = &tile[n * 288 + hd * 24];
  float w16[16];
  float mx = -1e30f;
#pragma unroll
  for (int m = 0; m < 16; ++m) {
    const float* ap = &anc[m * 96 + hd * 24];
    float s = 0.f;
#pragma unroll
    for (int d = 0; d < 24; ++d) s += (float)qp[d] * ap[d];
    s *= scale;
    w16[m] = s;
    mx = fmaxf(mx, s);
  }
  float sum = 0.f;
#pragma unroll
  for (int m = 0; m < 16; ++m) { w16[m] = __expf(w16[m] - mx); sum += w16[m]; }
  const float inv = 1.f / sum;
  float od[24];
#pragma unroll
  for (int d = 0; d < 24; ++d) od[d] = 0.f;
#pragma unroll
  for (int m = 0; m < 16; ++m) {
    const float pr = w16[m];
    const float* tp = &tmat[(hd * 16 + m) * 24];
#pragma unroll
    for (int d = 0; d < 24; ++d) od[d] += pr * tp[d];
  }
  const size_t tok = tokbase + (size_t)(wh * WSZ + (n >> 3)) * WW_ + (ww * WSZ + (n & 7));
#pragma unroll
  for (int d = 0; d < 24; ++d)
    out[tok * CC_ + 96 + hd * 24 + d] = (h16)(od[d] * inv);
}

// ---------------------------------------------------------------------------
// Utility kernels
// ---------------------------------------------------------------------------
__global__ void copy_f32(const float* __restrict__ in, float* __restrict__ out, int n4) {
  const int i = blockIdx.x * 256 + threadIdx.x;
  if (i < n4) ((float4*)out)[i] = ((const float4*)in)[i];
}
__global__ void f32_to_f16(const float* __restrict__ in, h16* __restrict__ out, int n) {
  const int i = blockIdx.x * 256 + threadIdx.x;
  if (i < n) out[i] = (h16)in[i];
}
// W[K][N] f32 -> Wt[N][K] f16
__global__ void wt_transpose(const float* __restrict__ W, h16* __restrict__ Wt, int K, int N) {
  const int idx = blockIdx.x * 256 + threadIdx.x;
  if (idx >= K * N) return;
  const int n = idx / K, k = idx % K;
  Wt[idx] = (h16)W[(size_t)k * N + n];
}
// conv_w [c_out][c_in][3][3] f32 -> Wt[c_out][tap*192+c_in] f16
__global__ void conv_wt(const float* __restrict__ W, h16* __restrict__ Wt) {
  const int idx = blockIdx.x * 256 + threadIdx.x;
  if (idx >= CC_ * 9 * CC_) return;
  const int n = idx / (9 * CC_), k = idx % (9 * CC_);
  const int tap = k / CC_, cin = k % CC_;
  Wt[idx] = (h16)W[((size_t)n * CC_ + cin) * 9 + tap];
}

// ---------------------------------------------------------------------------
extern "C" void kernel_launch(void* const* d_in, const int* in_sizes, int n_in,
                              void* d_out, int out_size, void* d_ws, size_t ws_size,
                              hipStream_t stream) {
  (void)in_sizes; (void)n_in; (void)out_size; (void)ws_size;
  const float* in_x    = (const float*)d_in[0];
  const float* n1g     = (const float*)d_in[1];
  const float* n1b     = (const float*)d_in[2];
  const float* qwwin   = (const float*)d_in[3];
  const float* qbwin   = (const float*)d_in[4];
  const float* qwstr   = (const float*)d_in[5];
  const float* qbstr   = (const float*)d_in[6];
  const float* projw   = (const float*)d_in[7];
  const float* projb   = (const float*)d_in[8];
  const float* rpbt    = (const float*)d_in[9];
  const float* n2g     = (const float*)d_in[10];
  const float* n2b     = (const float*)d_in[11];
  const float* fc1w    = (const float*)d_in[12];
  const float* fc1b    = (const float*)d_in[13];
  const float* fc2w    = (const float*)d_in[14];
  const float* fc2b    = (const float*)d_in[15];
  const float* convw   = (const float*)d_in[16];
  const float* convb   = (const float*)d_in[17];
  float* y_out = (float*)d_out;

  size_t off = 0;
  auto alloc = [&](size_t bytes) -> void* {
    off = (off + 255) & ~(size_t)255;
    void* p = (char*)d_ws + off;
    off += bytes;
    return p;
  };
  float* x       = (float*)alloc((size_t)TT_ * CC_ * 4);
  h16*   xn      = (h16*)  alloc((size_t)TT_ * CC_ * 2);
  h16*   qkvb    = (h16*)  alloc((size_t)TT_ * 3 * CH_ * 2);
  h16*   attnout = (h16*)  alloc((size_t)TT_ * CC_ * 2);
  h16*   hidden  = (h16*)  alloc((size_t)TT_ * 4 * CC_ * 2);
  h16*   w_qkv   = (h16*)  alloc((size_t)3 * CH_ * CH_ * 2);     // [288][96]
  h16*   w_proj  = (h16*)  alloc((size_t)CC_ * CC_ * 2);         // [192][192]
  h16*   w_fc1   = (h16*)  alloc((size_t)4 * CC_ * CC_ * 2);     // [768][192]
  h16*   w_fc2   = (h16*)  alloc((size_t)CC_ * 4 * CC_ * 2);     // [192][768]
  h16*   w_conv  = (h16*)  alloc((size_t)CC_ * 9 * CC_ * 2);     // [192][1728]

  const dim3 blk256(256);
  const int MB = TT_ / BM;  // 576

  copy_f32<<<dim3((TT_ * CC_ / 4 + 255) / 256), blk256, 0, stream>>>(in_x, x, TT_ * CC_ / 4);

  for (int i = 0; i < 4; ++i) {
    const int shift = (i % 2 == 0) ? (WSZ / 2) : 0;
    ln_kernel<<<dim3(TT_), dim3(192), 0, stream>>>(x, n1g + i * CC_, n1b + i * CC_, xn);
    wt_transpose<<<dim3((96 * 288 + 255) / 256), blk256, 0, stream>>>(
        qwwin + (size_t)i * 96 * 288, w_qkv, 96, 288);
    gemm_wmma<<<dim3(MB, 288 / BN), blk256, 0, stream>>>(
        xn, CC_, w_qkv, 96, qbwin + i * 288, nullptr, qkvb, nullptr, 288, EPI_F16);
    win_attn<<<dim3(NWIN), blk256, 0, stream>>>(qkvb, rpbt + i * 225 * 4, attnout, shift);
    wt_transpose<<<dim3((96 * 288 + 255) / 256), blk256, 0, stream>>>(
        qwstr + (size_t)i * 96 * 288, w_qkv, 96, 288);
    gemm_wmma<<<dim3(MB, 288 / BN), blk256, 0, stream>>>(
        xn + CH_, CC_, w_qkv, 96, qbstr + i * 288, nullptr, qkvb, nullptr, 288, EPI_F16);
    stripe_attn<<<dim3(NWIN), blk256, 0, stream>>>(qkvb, xn, attnout);
    wt_transpose<<<dim3((CC_ * CC_ + 255) / 256), blk256, 0, stream>>>(
        projw + (size_t)i * CC_ * CC_, w_proj, CC_, CC_);
    gemm_wmma<<<dim3(MB, CC_ / BN), blk256, 0, stream>>>(
        attnout, CC_, w_proj, CC_, projb + i * CC_, x, nullptr, x, CC_, EPI_RES_F32);
    ln_kernel<<<dim3(TT_), dim3(192), 0, stream>>>(x, n2g + i * CC_, n2b + i * CC_, xn);
    wt_transpose<<<dim3((CC_ * 4 * CC_ + 255) / 256), blk256, 0, stream>>>(
        fc1w + (size_t)i * CC_ * 4 * CC_, w_fc1, CC_, 4 * CC_);
    gemm_wmma<<<dim3(MB, 4 * CC_ / BN), blk256, 0, stream>>>(
        xn, CC_, w_fc1, CC_, fc1b + i * 4 * CC_, nullptr, hidden, nullptr, 4 * CC_, EPI_GELU_F16);
    wt_transpose<<<dim3((4 * CC_ * CC_ + 255) / 256), blk256, 0, stream>>>(
        fc2w + (size_t)i * 4 * CC_ * CC_, w_fc2, 4 * CC_, CC_);
    gemm_wmma<<<dim3(MB, CC_ / BN), blk256, 0, stream>>>(
        hidden, 4 * CC_, w_fc2, 4 * CC_, fc2b + i * CC_, x, nullptr, x, CC_, EPI_RES_F32);
  }

  f32_to_f16<<<dim3((TT_ * CC_ + 255) / 256), blk256, 0, stream>>>(x, xn, TT_ * CC_);
  conv_wt<<<dim3((CC_ * 9 * CC_ + 255) / 256), blk256, 0, stream>>>(convw, w_conv);
  conv_gemm<<<dim3(MB, CC_ / BN), blk256, 0, stream>>>(xn, w_conv, convb, in_x, y_out);
}
